// DeepSeekMoELayer_11690900980107
// MI455X (gfx1250) — matrix operand order repaired and checked by reading, not compiled, not run on hardware
//
#include <hip/hip_runtime.h>

// ---------------------------------------------------------------------------
// DeepSeek-MoE layer for MI455X (gfx1250, wave32, WMMA bf16 16x16x32).
// bf16 WMMA inputs, f32 accumulate, f32 output. Sparse top-2 routed experts.
// Round 2: software-pipelined K-loops (double-buffered B operands), LDS-staged
// X tile in the routed kernel, global_prefetch on the B stream, rcp-based silu.
// ---------------------------------------------------------------------------

typedef __attribute__((ext_vector_type(16))) __bf16 v16bf;
typedef __attribute__((ext_vector_type(8)))  __bf16 v8bf;
typedef __attribute__((ext_vector_type(8)))  float  v8f;

#define D_MODEL 1024
#define H_SHARED 2048
#define H_ROUTED 1024
#define N_EXPERTS 8

__device__ __forceinline__ v8f wmma_bf16(v16bf a, v16bf b, v8f c) {
  return __builtin_amdgcn_wmma_f32_16x16x32_bf16(false, a, false, b, (short)0,
                                                 c, false, false);
}

// A operand from row-major bf16 (global or LDS): lane l -> row l%16,
// half = l/16; two contiguous 16B loads at k0+half*8 and k0+half*8+16.
__device__ __forceinline__ v16bf load_a16(const __bf16* rowbase, int k0, int half) {
  const __bf16* p = rowbase + k0 + half * 8;
  v8bf lo = *(const v8bf*)(p);
  v8bf hi = *(const v8bf*)(p + 16);
  v16bf a;
#pragma unroll
  for (int i = 0; i < 8; ++i) { a[i] = lo[i]; a[i + 8] = hi[i]; }
  return a;
}

// B operand from a pre-swizzled 1KB tile: lane l reads 32 contiguous bytes.
__device__ __forceinline__ v16bf load_btile(const __bf16* tile, int lane) {
  const __bf16* p = tile + lane * 16;
  v8bf lo = *(const v8bf*)(p);
  v8bf hi = *(const v8bf*)(p + 8);
  v16bf b;
#pragma unroll
  for (int i = 0; i < 8; ++i) { b[i] = lo[i]; b[i + 8] = hi[i]; }
  return b;
}

__device__ __forceinline__ float fast_sigmoid(float x) {
  return __builtin_amdgcn_rcpf(1.f + __expf(-x));  // v_rcp_f32, no div expansion
}

// ---------------------------------------------------------------------------
// f32 -> bf16 row-major conversion (activations)
// ---------------------------------------------------------------------------
__global__ void __launch_bounds__(256) convert_bf16(const float* __restrict__ src,
                                                    __bf16* __restrict__ dst, long n) {
  long i = (long)blockIdx.x * 256 + threadIdx.x;
  long stride = (long)gridDim.x * 256;
  for (; i < n; i += stride) dst[i] = (__bf16)src[i];
}

// ---------------------------------------------------------------------------
// Pack row-major f32 weight [K][Nc] into WMMA B-operand tiles (1KB each):
// tileIdx = nt*(K/32)+kt  (contiguous along kt for a fixed column tile, which
// makes the GEMM-side B stream sequential and prefetch-friendly).
// ---------------------------------------------------------------------------
__global__ void __launch_bounds__(256) pack_b(const float* __restrict__ src,
                                              __bf16* __restrict__ dst,
                                              int K, int Nc) {
  long e = blockIdx.y;
  src += e * (long)K * Nc;
  dst += e * (long)K * Nc;
  int Ktiles = K >> 5;
  long tid = (long)blockIdx.x * 256 + threadIdx.x;
  long total = (long)K * Nc / 2;
  if (tid >= total) return;
  int v = (int)(tid & 7);
  int l = (int)((tid >> 3) & 31);
  long t = tid >> 8;
  int kt = (int)(t % Ktiles);
  int nt = (int)(t / Ktiles);
  int h = l >> 4;
  int n = nt * 16 + (l & 15);
  int k = kt * 32 + (v >> 2) * 16 + h * 8 + (v & 3) * 2;
  __bf16 a = (__bf16)src[(long)k * Nc + n];
  __bf16 b = (__bf16)src[(long)(k + 1) * Nc + n];
  long o = t * 512 + (long)l * 16 + v * 2;
  dst[o] = a;
  dst[o + 1] = b;
}

// ---------------------------------------------------------------------------
// Router affinities: one wave per (token, expert); s = sigmoid(x . w_e)
// ---------------------------------------------------------------------------
__global__ void __launch_bounds__(256) router_scores(const float* __restrict__ x,
                                                     const float* __restrict__ rw,
                                                     float* __restrict__ scores,
                                                     long N) {
  long gid = (long)blockIdx.x * 256 + threadIdx.x;
  long wid = gid >> 5;
  int lane = (int)(gid & 31);
  if (wid >= N * N_EXPERTS) return;
  long n = wid >> 3;
  int e = (int)(wid & 7);
  const float* xp = x + n * D_MODEL;
  const float* wp = rw + (long)e * D_MODEL;
  float s = 0.f;
  for (int i = lane; i < D_MODEL; i += 32) s += xp[i] * wp[i];
#pragma unroll
  for (int off = 16; off > 0; off >>= 1) s += __shfl_down(s, off);
  if (lane == 0) scores[wid] = 1.f / (1.f + __expf(-s));  // keep f32-exact-ish
}

// ---------------------------------------------------------------------------
// Top-2 selection + gate normalization -> comb[N][8]
// ---------------------------------------------------------------------------
__global__ void __launch_bounds__(256) router_comb(const float* __restrict__ scores,
                                                   const float* __restrict__ bias,
                                                   float* __restrict__ comb, long N) {
  long n = (long)blockIdx.x * 256 + threadIdx.x;
  if (n >= N) return;
  float s[N_EXPERTS], sel[N_EXPERTS];
#pragma unroll
  for (int e = 0; e < N_EXPERTS; ++e) {
    s[e] = scores[n * N_EXPERTS + e];
    sel[e] = s[e] + bias[e];
  }
  int i1 = 0;
#pragma unroll
  for (int e = 1; e < N_EXPERTS; ++e)
    if (sel[e] > sel[i1]) i1 = e;
  int i2 = (i1 == 0) ? 1 : 0;
#pragma unroll
  for (int e = 0; e < N_EXPERTS; ++e)
    if (e != i1 && sel[e] > sel[i2]) i2 = e;
  float d = s[i1] + s[i2];
  float g1, g2;
  if (d > 1e-9f) { g1 = s[i1] / (d + 1e-9f); g2 = s[i2] / (d + 1e-9f); }
  else           { g1 = 0.5f; g2 = 0.5f; }
#pragma unroll
  for (int e = 0; e < N_EXPERTS; ++e) comb[n * N_EXPERTS + e] = 0.f;
  comb[n * N_EXPERTS + i1] = g1;
  comb[n * N_EXPERTS + i2] = g2;
}

// ---------------------------------------------------------------------------
// Shared expert up: h = bf16( silu(x@sw1) * (x@sw3) )  [N][2048]
// Wave: 16 tokens x 64 cols for BOTH w1 and w3 (8 tiles). Double-buffered B.
// ---------------------------------------------------------------------------
__global__ void __launch_bounds__(256) gemm_shared_up(const __bf16* __restrict__ xb,
                                                      const __bf16* __restrict__ w1t,
                                                      const __bf16* __restrict__ w3t,
                                                      __bf16* __restrict__ hsh) {
  const int KT = D_MODEL / 32;  // 32
  int lane = threadIdx.x & 31;
  int wave = threadIdx.x >> 5;
  int half = lane >> 4;
  long mt = blockIdx.x;
  int colBase = blockIdx.y * 512 + wave * 64;
  const __bf16* arow = xb + (mt * 16 + (lane & 15)) * (long)D_MODEL;
  const __bf16* w1c = w1t + ((long)(colBase >> 4) * KT) * 512;
  const __bf16* w3c = w3t + ((long)(colBase >> 4) * KT) * 512;

  v8f acc1[4] = {v8f{}, v8f{}, v8f{}, v8f{}};
  v8f acc3[4] = {v8f{}, v8f{}, v8f{}, v8f{}};
  v16bf b0[8], b1[8];

  v16bf a0 = load_a16(arow, 0, half);
#pragma unroll
  for (int j = 0; j < 4; ++j) {
    b0[j]     = load_btile(w1c + (long)j * KT * 512, lane);
    b0[4 + j] = load_btile(w3c + (long)j * KT * 512, lane);
  }

  for (int kt = 0; kt < KT; kt += 2) {
    // issue loads for kt+1 (fills pipeline while kt's WMMAs execute)
    v16bf a1 = load_a16(arow, (kt + 1) * 32, half);
#pragma unroll
    for (int j = 0; j < 4; ++j) {
      const __bf16* p1 = w1c + (long)j * KT * 512 + (long)(kt + 1) * 512;
      const __bf16* p3 = w3c + (long)j * KT * 512 + (long)(kt + 1) * 512;
      b1[j]     = load_btile(p1, lane);
      b1[4 + j] = load_btile(p3, lane);
      __builtin_prefetch((const void*)(p1 + 4 * 512), 0, 1);  // global_prefetch
      __builtin_prefetch((const void*)(p3 + 4 * 512), 0, 1);
    }
#pragma unroll
    for (int j = 0; j < 4; ++j) {
      acc1[j] = wmma_bf16(a0, b0[j], acc1[j]);
      acc3[j] = wmma_bf16(a0, b0[4 + j], acc3[j]);
    }
    if (kt + 2 < KT) {
      a0 = load_a16(arow, (kt + 2) * 32, half);
#pragma unroll
      for (int j = 0; j < 4; ++j) {
        b0[j]     = load_btile(w1c + (long)j * KT * 512 + (long)(kt + 2) * 512, lane);
        b0[4 + j] = load_btile(w3c + (long)j * KT * 512 + (long)(kt + 2) * 512, lane);
      }
    }
#pragma unroll
    for (int j = 0; j < 4; ++j) {
      acc1[j] = wmma_bf16(a1, b1[j], acc1[j]);
      acc3[j] = wmma_bf16(a1, b1[4 + j], acc3[j]);
    }
  }

#pragma unroll
  for (int j = 0; j < 4; ++j) {
#pragma unroll
    for (int r = 0; r < 8; ++r) {
      float t1 = acc1[j][r];
      float t3 = acc3[j][r];
      float v = t1 * fast_sigmoid(t1) * t3;  // silu(t1)*t3, v_rcp_f32 path
      long row = mt * 16 + r + half * 8;
      int col = colBase + j * 16 + (lane & 15);
      hsh[row * H_SHARED + col] = (__bf16)v;
    }
  }
}

// ---------------------------------------------------------------------------
// Shared expert down: out = h @ sw2 (writes d_out f32).
// Wave: 16 tokens x 128 cols (8 tiles); block covers all 1024 columns.
// ---------------------------------------------------------------------------
__global__ void __launch_bounds__(256) gemm_shared_down(const __bf16* __restrict__ hsh,
                                                        const __bf16* __restrict__ w2t,
                                                        float* __restrict__ out) {
  const int KT = H_SHARED / 32;  // 64
  int lane = threadIdx.x & 31;
  int wave = threadIdx.x >> 5;
  int half = lane >> 4;
  long mt = blockIdx.x;
  int colBase = wave * 128;
  const __bf16* arow = hsh + (mt * 16 + (lane & 15)) * (long)H_SHARED;
  const __bf16* w2c = w2t + ((long)(colBase >> 4) * KT) * 512;

  v8f acc[8] = {v8f{}, v8f{}, v8f{}, v8f{}, v8f{}, v8f{}, v8f{}, v8f{}};
  v16bf b0[8], b1[8];

  v16bf a0 = load_a16(arow, 0, half);
#pragma unroll
  for (int j = 0; j < 8; ++j) b0[j] = load_btile(w2c + (long)j * KT * 512, lane);

  for (int kt = 0; kt < KT; kt += 2) {
    v16bf a1 = load_a16(arow, (kt + 1) * 32, half);
#pragma unroll
    for (int j = 0; j < 8; ++j) {
      const __bf16* p = w2c + (long)j * KT * 512 + (long)(kt + 1) * 512;
      b1[j] = load_btile(p, lane);
      __builtin_prefetch((const void*)(p + 4 * 512), 0, 1);
    }
#pragma unroll
    for (int j = 0; j < 8; ++j) acc[j] = wmma_bf16(a0, b0[j], acc[j]);
    if (kt + 2 < KT) {
      a0 = load_a16(arow, (kt + 2) * 32, half);
#pragma unroll
      for (int j = 0; j < 8; ++j)
        b0[j] = load_btile(w2c + (long)j * KT * 512 + (long)(kt + 2) * 512, lane);
    }
#pragma unroll
    for (int j = 0; j < 8; ++j) acc[j] = wmma_bf16(a1, b1[j], acc[j]);
  }

#pragma unroll
  for (int j = 0; j < 8; ++j) {
#pragma unroll
    for (int r = 0; r < 8; ++r) {
      long row = mt * 16 + r + half * 8;
      int col = colBase + j * 16 + (lane & 15);
      out[row * D_MODEL + col] = acc[j][r];
    }
  }
}

// ---------------------------------------------------------------------------
// Routed experts: per 16-token tile, loop experts; skip when comb all-zero.
// X tile staged once in LDS (reused 8x per selected expert); h tile in LDS.
// Stage 1: h = gelu(X @ ew1[e]); Stage 2: out += comb .* (h @ ew2[e]).
// ---------------------------------------------------------------------------
__global__ void __launch_bounds__(256) gemm_routed(const __bf16* __restrict__ xb,
                                                   const __bf16* __restrict__ ew1t,
                                                   const __bf16* __restrict__ ew2t,
                                                   const float* __restrict__ comb,
                                                   float* __restrict__ out) {
  __shared__ __align__(16) __bf16 xlds[16 * D_MODEL];   // 32 KB
  __shared__ __align__(16) __bf16 h16[16 * H_ROUTED];   // 32 KB
  const int KT = 32;  // both GEMMs have K=1024
  int lane = threadIdx.x & 31;
  int wave = threadIdx.x >> 5;
  int half = lane >> 4;
  long mt = blockIdx.x;
  int colBase = wave * 128;

  // Cooperative copy: X tile (16 x 1024 bf16 = 32 KB) global -> LDS, once.
  {
    const v8bf* src = (const v8bf*)(xb + mt * 16 * (long)D_MODEL);
    v8bf* dst = (v8bf*)xlds;
#pragma unroll
    for (int i = 0; i < (16 * D_MODEL / 8) / 256; ++i)
      dst[threadIdx.x + i * 256] = src[threadIdx.x + i * 256];
  }
  __syncthreads();

  const __bf16* arow = xlds + (lane & 15) * D_MODEL;
  const __bf16* hrow = h16 + (lane & 15) * H_ROUTED;

  for (int e = 0; e < N_EXPERTS; ++e) {
    float cv = comb[(mt * 16 + (lane & 15)) * N_EXPERTS + e];
    if (__ballot(cv > 0.f) == 0ull) continue;  // block-uniform skip
    __syncthreads();  // previous iteration's LDS readers done

    // ---- stage 1: h(16x1024) = gelu(X @ ew1[e]) ----
    {
      const __bf16* w1c =
          ew1t + (long)e * D_MODEL * H_ROUTED + ((long)(colBase >> 4) * KT) * 512;
      v8f acc[8] = {v8f{}, v8f{}, v8f{}, v8f{}, v8f{}, v8f{}, v8f{}, v8f{}};
      v16bf b0[8], b1[8];
#pragma unroll
      for (int j = 0; j < 8; ++j) b0[j] = load_btile(w1c + (long)j * KT * 512, lane);

      for (int kt = 0; kt < KT; kt += 2) {
#pragma unroll
        for (int j = 0; j < 8; ++j) {
          const __bf16* p = w1c + (long)j * KT * 512 + (long)(kt + 1) * 512;
          b1[j] = load_btile(p, lane);
          __builtin_prefetch((const void*)(p + 4 * 512), 0, 1);
        }
        v16bf a0 = load_a16(arow, kt * 32, half);  // ds_load_b128 (cheap)
#pragma unroll
        for (int j = 0; j < 8; ++j) acc[j] = wmma_bf16(a0, b0[j], acc[j]);
        if (kt + 2 < KT) {
#pragma unroll
          for (int j = 0; j < 8; ++j)
            b0[j] = load_btile(w1c + (long)j * KT * 512 + (long)(kt + 2) * 512, lane);
        }
        v16bf a1 = load_a16(arow, (kt + 1) * 32, half);
#pragma unroll
        for (int j = 0; j < 8; ++j) acc[j] = wmma_bf16(a1, b1[j], acc[j]);
      }
#pragma unroll
      for (int j = 0; j < 8; ++j) {
#pragma unroll
        for (int r = 0; r < 8; ++r) {
          float v = acc[j][r];
          float g = 0.5f * v * (1.f + erff(v * 0.70710678f));  // exact GELU
          int row = r + half * 8;
          int col = colBase + j * 16 + (lane & 15);
          h16[row * H_ROUTED + col] = (__bf16)g;
        }
      }
    }
    __syncthreads();

    // ---- stage 2: out += comb .* (h @ ew2[e]) ----
    {
      const __bf16* w2c =
          ew2t + (long)e * H_ROUTED * D_MODEL + ((long)(colBase >> 4) * KT) * 512;
      v8f acc[8] = {v8f{}, v8f{}, v8f{}, v8f{}, v8f{}, v8f{}, v8f{}, v8f{}};
      v16bf b0[8], b1[8];
#pragma unroll
      for (int j = 0; j < 8; ++j) b0[j] = load_btile(w2c + (long)j * KT * 512, lane);

      for (int kt = 0; kt < KT; kt += 2) {
#pragma unroll
        for (int j = 0; j < 8; ++j) {
          const __bf16* p = w2c + (long)j * KT * 512 + (long)(kt + 1) * 512;
          b1[j] = load_btile(p, lane);
          __builtin_prefetch((const void*)(p + 4 * 512), 0, 1);
        }
        v16bf a0 = load_a16(hrow, kt * 32, half);
#pragma unroll
        for (int j = 0; j < 8; ++j) acc[j] = wmma_bf16(a0, b0[j], acc[j]);
        if (kt + 2 < KT) {
#pragma unroll
          for (int j = 0; j < 8; ++j)
            b0[j] = load_btile(w2c + (long)j * KT * 512 + (long)(kt + 2) * 512, lane);
        }
        v16bf a1 = load_a16(hrow, (kt + 1) * 32, half);
#pragma unroll
        for (int j = 0; j < 8; ++j) acc[j] = wmma_bf16(a1, b1[j], acc[j]);
      }
#pragma unroll
      for (int j = 0; j < 8; ++j) {
#pragma unroll
        for (int r = 0; r < 8; ++r) {
          int row = r + half * 8;
          float c = comb[(mt * 16 + row) * N_EXPERTS + e];
          int col = colBase + j * 16 + (lane & 15);
          long o = (mt * 16 + row) * (long)D_MODEL + col;
          out[o] += c * acc[j][r];  // unique (row,col) per lane elem; serial in e
        }
      }
    }
  }
}

// ---------------------------------------------------------------------------
extern "C" void kernel_launch(void* const* d_in, const int* in_sizes, int n_in,
                              void* d_out, int out_size, void* d_ws, size_t ws_size,
                              hipStream_t stream) {
  const float* x   = (const float*)d_in[0];
  const float* rw  = (const float*)d_in[1];
  const float* rb  = (const float*)d_in[2];
  const float* sw1 = (const float*)d_in[3];
  const float* sw3 = (const float*)d_in[4];
  const float* sw2 = (const float*)d_in[5];
  const float* ew1 = (const float*)d_in[6];
  const float* ew2 = (const float*)d_in[7];
  float* out = (float*)d_out;

  long N = (long)in_sizes[0] / D_MODEL;  // 4096 tokens

  char* ws = (char*)d_ws;
  __bf16* xb   = (__bf16*)ws; ws += N * D_MODEL * 2;
  __bf16* hsh  = (__bf16*)ws; ws += N * H_SHARED * 2;
  __bf16* sw1t = (__bf16*)ws; ws += (long)D_MODEL * H_SHARED * 2;
  __bf16* sw3t = (__bf16*)ws; ws += (long)D_MODEL * H_SHARED * 2;
  __bf16* sw2t = (__bf16*)ws; ws += (long)H_SHARED * D_MODEL * 2;
  __bf16* ew1t = (__bf16*)ws; ws += (long)N_EXPERTS * D_MODEL * H_ROUTED * 2;
  __bf16* ew2t = (__bf16*)ws; ws += (long)N_EXPERTS * H_ROUTED * D_MODEL * 2;
  float* scores = (float*)ws; ws += N * N_EXPERTS * 4;
  float* comb   = (float*)ws; ws += N * N_EXPERTS * 4;

  // 1) Precision conversion + weight swizzle (one HBM/L2 pass, ~100 MB)
  {
    long n = N * D_MODEL;
    convert_bf16<<<dim3((unsigned)((n + 255) / 256)), 256, 0, stream>>>(x, xb, n);
  }
  {
    unsigned blk = (unsigned)(((long)D_MODEL * H_SHARED / 2 + 255) / 256);
    pack_b<<<dim3(blk, 1), 256, 0, stream>>>(sw1, sw1t, D_MODEL, H_SHARED);
    pack_b<<<dim3(blk, 1), 256, 0, stream>>>(sw3, sw3t, D_MODEL, H_SHARED);
  }
  {
    unsigned blk = (unsigned)(((long)H_SHARED * D_MODEL / 2 + 255) / 256);
    pack_b<<<dim3(blk, 1), 256, 0, stream>>>(sw2, sw2t, H_SHARED, D_MODEL);
  }
  {
    unsigned blk = (unsigned)(((long)D_MODEL * H_ROUTED / 2 + 255) / 256);
    pack_b<<<dim3(blk, N_EXPERTS), 256, 0, stream>>>(ew1, ew1t, D_MODEL, H_ROUTED);
    pack_b<<<dim3(blk, N_EXPERTS), 256, 0, stream>>>(ew2, ew2t, H_ROUTED, D_MODEL);
  }

  // 2) Router (f32, as reference)
  router_scores<<<dim3((unsigned)N), 256, 0, stream>>>(x, rw, scores, N);
  router_comb<<<dim3((unsigned)((N + 255) / 256)), 256, 0, stream>>>(scores, rb, comb, N);

  // 3) Shared expert SwiGLU
  gemm_shared_up<<<dim3((unsigned)(N / 16), H_SHARED / 512), 256, 0, stream>>>(
      xb, sw1t, sw3t, hsh);
  gemm_shared_down<<<dim3((unsigned)(N / 16)), 256, 0, stream>>>(hsh, sw2t, out);

  // 4) Routed experts (sparse top-2 equivalent), accumulates into d_out
  gemm_routed<<<dim3((unsigned)(N / 16)), 256, 0, stream>>>(xb, ew1t, ew2t, comb, out);
}
